// NCELoss_90039694393513
// MI455X (gfx1250) — compile-verified, matrix operand order checked by reading, not run on hardware
//
#include <hip/hip_runtime.h>
#include <hip/hip_bf16.h>

// ---------- problem constants (from reference setup_inputs) ----------
#define B_SZ 4096
#define E_SZ 1024
#define S_SZ 4096

typedef __bf16 bf16_t;
typedef bf16_t v16bf __attribute__((ext_vector_type(16)));
typedef bf16_t v8bf  __attribute__((ext_vector_type(8)));
typedef bf16_t v4bf  __attribute__((ext_vector_type(4)));
typedef float  v8f   __attribute__((ext_vector_type(8)));

// ---------------------------------------------------------------------
// Fragment loader for 16-bit WMMA operands (A: 16xK, B: Kx16, both stored
// row-major along K).  Per ISA 7.12.2 layout for 16-bit 16x32 fragments:
//   lanes 0-15  hold K = [k .. k+7] and [k+16 .. k+23]   (row/col = lane)
//   lanes 16-31 hold K = [k+8..k+15] and [k+24 .. k+31]  (row/col = lane-16)
// Caller passes rowp = &M[row][0] and k0 = kstep + ((lane>>4)<<3).
// Both chunks are 16-byte aligned -> global_load_b128.
// ---------------------------------------------------------------------
__device__ __forceinline__ v16bf load_frag16(const bf16_t* __restrict__ rowp, int k0) {
    v8bf lo = *(const v8bf*)(rowp + k0);
    v8bf hi = *(const v8bf*)(rowp + k0 + 16);
    v16bf r;
#pragma unroll
    for (int i = 0; i < 8; ++i) { r[i] = lo[i]; r[i + 8] = hi[i]; }
    return r;
}

__device__ __forceinline__ float wave_reduce_add(float v) {
#pragma unroll
    for (int off = 16; off > 0; off >>= 1) v += __shfl_xor(v, off, 32);
    return v;
}

__device__ __forceinline__ float log_sigmoid(float x) {
    // stable: min(x,0) - log1p(exp(-|x|))
    return fminf(x, 0.0f) - log1pf(expf(-fabsf(x)));
}

// ---------------------------------------------------------------------
// Kernel 0: zero the two scalar accumulators
// ---------------------------------------------------------------------
__global__ void zero_accum(float* acc) {
    if (threadIdx.x < 2) acc[threadIdx.x] = 0.0f;
}

// ---------------------------------------------------------------------
// Kernel 1: per-row L2 norms, true-term, bf16 normalized input.
// One block (256 threads, 8 waves) per row; each thread owns 4 elements.
// ---------------------------------------------------------------------
__global__ __launch_bounds__(256) void prep_rows(const float* __restrict__ in,
                                                 const float* __restrict__ tg,
                                                 bf16_t* __restrict__ inorm_bf,
                                                 float* __restrict__ true_acc) {
    __shared__ float red[3][8];
    const int row = blockIdx.x;
    const int tid = threadIdx.x;
    const float4 xv = ((const float4*)(in + (size_t)row * E_SZ))[tid];
    const float4 tv = ((const float4*)(tg + (size_t)row * E_SZ))[tid];

    float sxx = xv.x * xv.x + xv.y * xv.y + xv.z * xv.z + xv.w * xv.w;
    float stt = tv.x * tv.x + tv.y * tv.y + tv.z * tv.z + tv.w * tv.w;
    float sxt = xv.x * tv.x + xv.y * tv.y + xv.z * tv.z + xv.w * tv.w;

    sxx = wave_reduce_add(sxx);
    stt = wave_reduce_add(stt);
    sxt = wave_reduce_add(sxt);

    const int lane = tid & 31, wave = tid >> 5;
    if (lane == 0) { red[0][wave] = sxx; red[1][wave] = stt; red[2][wave] = sxt; }
    __syncthreads();

    float s0 = 0.f, s1 = 0.f, s2 = 0.f;
#pragma unroll
    for (int w = 0; w < 8; ++w) { s0 += red[0][w]; s1 += red[1][w]; s2 += red[2][w]; }

    const float nin = fmaxf(sqrtf(s0), 1e-12f);
    const float ntg = fmaxf(sqrtf(s1), 1e-12f);
    const float rn  = 1.0f / nin;

    v4bf o;
    o[0] = (bf16_t)(xv.x * rn); o[1] = (bf16_t)(xv.y * rn);
    o[2] = (bf16_t)(xv.z * rn); o[3] = (bf16_t)(xv.w * rn);
    *(v4bf*)(inorm_bf + (size_t)row * E_SZ + tid * 4) = o;

    if (tid == 0) {
        float d = s2 / (nin * ntg);                // cosine similarity
        atomicAdd(true_acc, log_sigmoid(d));       // sum of true_log_prob
    }
}

// ---------------------------------------------------------------------
// Kernel 2: Wt[n][k] = bf16(W[k][n])  (LDS-tiled 32x32 transpose)
// ---------------------------------------------------------------------
__global__ __launch_bounds__(256) void transpose_w_bf16(const float* __restrict__ W,
                                                        bf16_t* __restrict__ Wt) {
    __shared__ float tile[32][33];
    const int bx = blockIdx.x * 32;  // n base on input
    const int by = blockIdx.y * 32;  // k base on input
    const int tx = threadIdx.x;      // 0..31
    const int ty = threadIdx.y;      // 0..7
#pragma unroll
    for (int j = 0; j < 32; j += 8)
        tile[ty + j][tx] = W[(size_t)(by + ty + j) * E_SZ + (bx + tx)];
    __syncthreads();
#pragma unroll
    for (int j = 0; j < 32; j += 8)
        Wt[(size_t)(bx + ty + j) * E_SZ + (by + tx)] = (bf16_t)tile[tx][ty + j];
}

// ---------------------------------------------------------------------
// Kernel 3: float -> bf16 elementwise convert (noise matrix)
// ---------------------------------------------------------------------
__global__ __launch_bounds__(256) void convert_bf16(const float* __restrict__ src,
                                                    bf16_t* __restrict__ dst) {
    const size_t i = (size_t)blockIdx.x * blockDim.x + threadIdx.x;
    const float4 v = ((const float4*)src)[i];
    v4bf o;
    o[0] = (bf16_t)v.x; o[1] = (bf16_t)v.y; o[2] = (bf16_t)v.z; o[3] = (bf16_t)v.w;
    ((v4bf*)dst)[i] = o;
}

// ---------------------------------------------------------------------
// Kernel 4: GEMM1  P[m][n] = sum_k A[m][k] * Bm[n][k]   (NT, bf16 in, bf16 out)
// Block = 8 waves; block tile = 32(M) x 512(N).
// Each wave: 32x64 register tile = 8 accumulators; per k-step:
//   2 A-frags + 4 B-frags (12 b128 loads) feed 8 WMMAs  (1.5 loads/WMMA).
// ---------------------------------------------------------------------
__global__ __launch_bounds__(256) void gemm_nt_store_bf16(const bf16_t* __restrict__ A,
                                                          const bf16_t* __restrict__ Bm,
                                                          bf16_t* __restrict__ Out,
                                                          int K, int N) {
    const int lane = threadIdx.x & 31;
    const int wave = threadIdx.x >> 5;
    const int m0 = blockIdx.x * 32;
    const int n0 = blockIdx.y * 512 + wave * 64;
    const int r    = lane & 15;
    const int ksel = (lane >> 4) << 3;

    const bf16_t* arow0 = A + (size_t)(m0 + r) * K;
    const bf16_t* arow1 = A + (size_t)(m0 + 16 + r) * K;
    v8f acc[2][4] = {};
    for (int k = 0; k < K; k += 32) {
        const v16bf af0 = load_frag16(arow0, k + ksel);
        const v16bf af1 = load_frag16(arow1, k + ksel);
#pragma unroll
        for (int s = 0; s < 4; ++s) {
            const v16bf bf = load_frag16(Bm + (size_t)(n0 + s * 16 + r) * K, k + ksel);
            acc[0][s] = __builtin_amdgcn_wmma_f32_16x16x32_bf16(
                false, af0, false, bf, (short)0, acc[0][s], false, false);
            acc[1][s] = __builtin_amdgcn_wmma_f32_16x16x32_bf16(
                false, af1, false, bf, (short)0, acc[1][s], false, false);
        }
    }
    // C/D layout: vgpr i -> m = mbase + i + (lane>=16 ? 8 : 0); n = n0 + s*16 + (lane&15)
#pragma unroll
    for (int t = 0; t < 2; ++t) {
        const int mlo = m0 + t * 16 + ((lane >> 4) << 3);
#pragma unroll
        for (int s = 0; s < 4; ++s) {
            const int n = n0 + s * 16 + r;
#pragma unroll
            for (int i = 0; i < 8; ++i)
                Out[(size_t)(mlo + i) * N + n] = (bf16_t)acc[t][s][i];
        }
    }
}

// ---------------------------------------------------------------------
// Kernel 5: GEMM2 + fused epilogue (same 32x64 wave tile).
// scores[m][n] = sum_k P[m][k]*noise[n][k] + bias[n];
// accumulate sum of log_sigmoid(-score) = -softplus(score) into noise_acc.
// ---------------------------------------------------------------------
__global__ __launch_bounds__(256) void gemm_nt_logsig_reduce(const bf16_t* __restrict__ A,
                                                             const bf16_t* __restrict__ Bm,
                                                             const float* __restrict__ bias,
                                                             float* __restrict__ noise_acc,
                                                             int K) {
    const int lane = threadIdx.x & 31;
    const int wave = threadIdx.x >> 5;
    const int m0 = blockIdx.x * 32;
    const int n0 = blockIdx.y * 512 + wave * 64;
    const int r    = lane & 15;
    const int ksel = (lane >> 4) << 3;

    const bf16_t* arow0 = A + (size_t)(m0 + r) * K;
    const bf16_t* arow1 = A + (size_t)(m0 + 16 + r) * K;
    v8f acc[2][4] = {};
    for (int k = 0; k < K; k += 32) {
        const v16bf af0 = load_frag16(arow0, k + ksel);
        const v16bf af1 = load_frag16(arow1, k + ksel);
#pragma unroll
        for (int s = 0; s < 4; ++s) {
            const v16bf bf = load_frag16(Bm + (size_t)(n0 + s * 16 + r) * K, k + ksel);
            acc[0][s] = __builtin_amdgcn_wmma_f32_16x16x32_bf16(
                false, af0, false, bf, (short)0, acc[0][s], false, false);
            acc[1][s] = __builtin_amdgcn_wmma_f32_16x16x32_bf16(
                false, af1, false, bf, (short)0, acc[1][s], false, false);
        }
    }

    float local = 0.0f;
#pragma unroll
    for (int s = 0; s < 4; ++s) {
        const float bb = bias[n0 + s * 16 + r];
#pragma unroll
        for (int t = 0; t < 2; ++t) {
#pragma unroll
            for (int i = 0; i < 8; ++i) {
                const float x = acc[t][s][i] + bb;
                // log_sigmoid(-x) = -softplus(x), numerically stable
                local -= fmaxf(x, 0.0f) + log1pf(expf(-fabsf(x)));
            }
        }
    }
    local = wave_reduce_add(local);
    if (lane == 0) atomicAdd(noise_acc, local);
}

// ---------------------------------------------------------------------
// Kernel 6: finalize scalar loss
// ---------------------------------------------------------------------
__global__ void finalize(const float* __restrict__ acc, float* __restrict__ out) {
    out[0] = -(acc[0] + acc[1]) * (1.0f / (float)B_SZ);
}

// ---------------------------------------------------------------------
extern "C" void kernel_launch(void* const* d_in, const int* in_sizes, int n_in,
                              void* d_out, int out_size, void* d_ws, size_t ws_size,
                              hipStream_t stream) {
    const float* input  = (const float*)d_in[0];  // [B,E]
    const float* target = (const float*)d_in[1];  // [B,E]
    const float* W      = (const float*)d_in[2];  // [E,E]
    const float* bias   = (const float*)d_in[3];  // [S]
    const float* noise  = (const float*)d_in[4];  // [S,E]
    float* out = (float*)d_out;

    // workspace layout (bytes)
    char* ws = (char*)d_ws;
    bf16_t* inorm_bf = (bf16_t*)(ws);                                   // B*E*2  = 8388608
    bf16_t* Wt_bf    = (bf16_t*)(ws + 8388608);                         // E*E*2  = 2097152
    bf16_t* noise_bf = (bf16_t*)(ws + 8388608 + 2097152);               // S*E*2  = 8388608
    bf16_t* P_bf     = (bf16_t*)(ws + 8388608 + 2097152 + 8388608);     // B*E*2  = 8388608
    float*  acc      = (float*)(ws + 8388608 + 2097152 + 8388608 + 8388608); // 2 floats

    zero_accum<<<1, 32, 0, stream>>>(acc);

    // 1) row norms + true term + bf16 normalized input
    prep_rows<<<B_SZ, 256, 0, stream>>>(input, target, inorm_bf, acc + 0);

    // 2) W -> Wt (bf16, transposed so GEMM1 is NT form)
    transpose_w_bf16<<<dim3(E_SZ / 32, E_SZ / 32), dim3(32, 8), 0, stream>>>(W, Wt_bf);

    // 3) noise -> bf16
    convert_bf16<<<(S_SZ * E_SZ / 4) / 256, 256, 0, stream>>>(noise, noise_bf);

    // 4) P = input_norm @ W   (as inorm_bf[m][k] * Wt_bf[n][k])
    gemm_nt_store_bf16<<<dim3(B_SZ / 32, E_SZ / 512), 256, 0, stream>>>(
        inorm_bf, Wt_bf, P_bf, E_SZ, E_SZ);

    // 5) scores = P @ noise^T + b, fused log_sigmoid(-x) grand reduction
    gemm_nt_logsig_reduce<<<dim3(B_SZ / 32, S_SZ / 512), 256, 0, stream>>>(
        P_bf, noise_bf, bias, acc + 1, E_SZ);

    // 6) loss
    finalize<<<1, 1, 0, stream>>>(acc, out);
}
